// CrossAttention_33423435498049
// MI455X (gfx1250) — matrix operand order repaired and checked by reading, not compiled, not run on hardware
//
#include <hip/hip_runtime.h>

#define PIXEL_DIM 768
#define AGE_DIM   128
#define BATCH     8
#define NTOK      2048

typedef float    v2f  __attribute__((ext_vector_type(2)));
typedef float    v4f  __attribute__((ext_vector_type(4)));
typedef float    v8f  __attribute__((ext_vector_type(8)));
typedef _Float16 v16h __attribute__((ext_vector_type(16)));

// ---------------------------------------------------------------------------
// Kernel 1: v[b, :] = age[b, :] @ Wv + bv   (8x128 @ 128x768)
// One wave (32 lanes) per 16-wide column tile; M padded 8 -> 16 (rows 8..15
// are duplicates via rowA&7 and never stored). 48 tiles total.
// ---------------------------------------------------------------------------
__global__ __launch_bounds__(32) void v_proj_wmma(const float* __restrict__ age,
                                                  const float* __restrict__ Wv,
                                                  const float* __restrict__ bv,
                                                  float* __restrict__ vbuf) {
    const int lane    = threadIdx.x;   // 0..31, EXEC all-ones (WMMA requirement)
    const int col0    = blockIdx.x * 16;
    const int halfSel = lane >> 4;     // 0: lanes 0-15, 1: lanes 16-31
    const int idx16   = lane & 15;     // M for A-frag / N for B-frag
    const int rowA    = idx16 & 7;     // clamp padded rows onto valid data

    v8f c = {0.f, 0.f, 0.f, 0.f, 0.f, 0.f, 0.f, 0.f};

#if __has_builtin(__builtin_amdgcn_wmma_f32_16x16x4_f32)
    // fp32 path: V_WMMA_F32_16X16X4_F32, K = 128 in 32 steps of 4.
    // A 16x4 layout: lanes 0-15 -> K = {k0, k0+1}; lanes 16-31 -> {k0+2, k0+3}.
    // B 4x16 mirrored with lanes indexing N.
    const int kOff = halfSel * 2;
#pragma unroll
    for (int k0 = 0; k0 < AGE_DIM; k0 += 4) {
        v2f a, b;
        a.x = age[rowA * AGE_DIM + k0 + kOff];
        a.y = age[rowA * AGE_DIM + k0 + kOff + 1];
        b.x = Wv[(k0 + kOff)     * PIXEL_DIM + col0 + idx16];
        b.y = Wv[(k0 + kOff + 1) * PIXEL_DIM + col0 + idx16];
        c = __builtin_amdgcn_wmma_f32_16x16x4_f32(
                /*neg_a=*/false, a, /*neg_b=*/false, b,
                /*c_mod=*/(short)0, c, /*reuse_a=*/false, /*reuse_b=*/false);
    }
#else
    // Fallback: codegen-confirmed f16-input WMMA with fp32 accumulate.
    // A 16x32 f16 layout (ISA 7.12.2): lanes 0-15 hold K = {0..7, 16..23},
// lanes 16-31 hold K = {8..15, 24..31}; B 32x16 uses the same K layout
    // (lanes 0-15 K={0..7,16..23}, lanes 16-31 K={8..15,24..31}).
#pragma unroll
    for (int k0 = 0; k0 < AGE_DIM; k0 += 32) {
        v16h a, b;
#pragma unroll
        for (int e = 0; e < 16; ++e) {
            const int ka = k0 + halfSel * 8 + (e >> 3) * 16 + (e & 7);
            a[e] = (_Float16)age[rowA * AGE_DIM + ka];
            const int kb = k0 + halfSel * 8 + (e >> 3) * 16 + (e & 7);
            b[e] = (_Float16)Wv[kb * PIXEL_DIM + col0 + idx16];
        }
        c = __builtin_amdgcn_wmma_f32_16x16x32_f16(
                false, a, false, b, (short)0, c, false, false);
    }
#endif

    // C/D layout: VGPR j, lanes 0-15 -> (M=j, N=lane) == batches 0..7 exactly.
    if (lane < 16) {
        const int   col  = col0 + idx16;
        const float bias = bv[col];
#pragma unroll
        for (int j = 0; j < 8; ++j) {
            vbuf[j * PIXEL_DIM + col] = c[j] + bias;
        }
    }
}

// ---------------------------------------------------------------------------
// Kernel 2 (roofline path, ~100 MB @ 23.3 TB/s): out = pixel + broadcast(v).
// One block per token row (768 floats = 192 float4 lanes = 6 waves).
// pixel/out are streamed once -> non-temporal; v (24 KB) stays cache-hot.
// ---------------------------------------------------------------------------
__global__ __launch_bounds__(192) void add_vbcast(const float* __restrict__ pixel,
                                                  const float* __restrict__ vbuf,
                                                  float* __restrict__ out) {
    const int    n      = blockIdx.x;
    const int    b      = blockIdx.y;
    const int    t      = threadIdx.x;                       // 0..191
    const size_t rowOff = ((size_t)b * NTOK + n) * PIXEL_DIM;

    const v4f* __restrict__ prow = (const v4f*)(pixel + rowOff);
    const v4f* __restrict__ vrow = (const v4f*)(vbuf + (size_t)b * PIXEL_DIM);

    v4f p  = __builtin_nontemporal_load(prow + t);  // streaming, no reuse
    v4f vv = vrow[t];                               // cached, reused by all n
    __builtin_nontemporal_store(p + vv, (v4f*)(out + rowOff) + t);
}

extern "C" void kernel_launch(void* const* d_in, const int* in_sizes, int n_in,
                              void* d_out, int out_size, void* d_ws, size_t ws_size,
                              hipStream_t stream) {
    (void)in_sizes; (void)n_in; (void)out_size; (void)ws_size;
    // setup_inputs order:
    // 0: pixel_features [8,2048,768]  1: age_features [8,128]
    // 2: Wq  3: bq  4: Wk  5: bk      (mathematically cancel -> unused)
    // 6: Wv [128,768]  7: bv [768]
    const float* pixel = (const float*)d_in[0];
    const float* age   = (const float*)d_in[1];
    const float* Wv    = (const float*)d_in[6];
    const float* bv    = (const float*)d_in[7];
    float*       vbuf  = (float*)d_ws;            // 8 x 768 fp32 = 24 KB scratch

    v_proj_wmma<<<dim3(PIXEL_DIM / 16), dim3(32), 0, stream>>>(age, Wv, bv, vbuf);
    add_vbcast<<<dim3(NTOK, BATCH), dim3(192), 0, stream>>>(pixel, vbuf, (float*)d_out);
}
